// InstanceAttentionModule_84731114815736
// MI455X (gfx1250) — compile-verified, matrix operand order checked by reading, not run on hardware
//
#include <hip/hip_runtime.h>
#include <hip/hip_bf16.h>

#define BATCH 32
#define HW    1024
#define CHN   256
#define FD    64
#define WPB   8     // waves per block
#define TJ    32    // j-tile (K chunk for the P*V WMMA)
#define NJ    (HW / TJ)

typedef __attribute__((ext_vector_type(16))) __bf16 v16bf;
typedef __attribute__((ext_vector_type(8)))  float  v8f;
typedef __attribute__((ext_vector_type(4))) unsigned int u32x4;
typedef __attribute__((ext_vector_type(8))) int i32x8;
typedef __attribute__((ext_vector_type(4))) int i32x4;

// ---- fragment loaders (CDNA5 16-bit A/B layout, wave32) ---------------------
// lane: row/col = lane&15 ; kb = 8*(lane>>4)
// elements 0..7  <- src[kb .. kb+7]
// elements 8..15 <- src[16+kb .. 16+kb+7]
__device__ __forceinline__ v16bf frag_from_f32(const float* __restrict__ rowk, int kb) {
  v16bf r;
#pragma unroll
  for (int e = 0; e < 8; ++e) r[e] = (__bf16)rowk[kb + e];
#pragma unroll
  for (int e = 0; e < 8; ++e) r[8 + e] = (__bf16)rowk[16 + kb + e];
  return r;
}
__device__ __forceinline__ v16bf frag_from_bf16(const __bf16* rowk, int kb) {
  v16bf r;
#pragma unroll
  for (int e = 0; e < 8; ++e) r[e] = rowk[kb + e];
#pragma unroll
  for (int e = 0; e < 8; ++e) r[8 + e] = rowk[16 + kb + e];
  return r;
}

// ---- Tensor Data Mover: 2D tile (16-bit elements) global -> LDS -------------
// D# per CDNA5 ISA: group0 = {ctl, lds_addr,
// global_addr lo, global_addr hi | type=2}; group1 = {data_size=2B, tensor
// dims, tile dims, dim0 stride}; remaining descriptor words zero (2D tensor,
// no cluster multicast, no gather). Tracked with TENSORcnt; wave-level op
// (EXEC ignored), issued from uniform control flow.
// NOTE: this toolchain exposes the 6-arg builtin
//   (u32x4, i32x8, i32x4, i32x4, i32x8, i32 cpol).
__device__ __forceinline__ void tdm_load_2d_bf16(const void* gaddr, unsigned lds_addr,
                                                 unsigned tile_x, unsigned tile_y,
                                                 unsigned tensor_x, unsigned tensor_y,
                                                 unsigned stride_x) {
  const unsigned long long ga = (unsigned long long)gaddr;
  u32x4 g0;
  g0[0] = 1u;                                              // count=1, user mode
  g0[1] = lds_addr;                                        // LDS byte address
  g0[2] = (unsigned)ga;                                    // global addr [95:64]
  g0[3] = (unsigned)((ga >> 32) & 0x1FFFFFFull) | (2u << 30); // addr hi | type=2
  i32x8 g1;
  g1[0] = (int)(1u << 16);                                 // data_size = 2 bytes
  g1[1] = (int)((tensor_x & 0xFFFFu) << 16);               // tensor_dim0 lo16
  g1[2] = (int)((tensor_x >> 16) | ((tensor_y & 0xFFFFu) << 16)); // dim0 hi | dim1 lo
  g1[3] = (int)((tensor_y >> 16) | (tile_x << 16));        // dim1 hi | tile_dim0
  g1[4] = (int)tile_y;                                     // tile_dim1 (tile_dim2=0)
  g1[5] = (int)stride_x;                                   // tensor_dim0_stride lo32
  g1[6] = 0;
  g1[7] = 0;
  const i32x4 gz4 = {0, 0, 0, 0};
  const i32x8 gz8 = {0, 0, 0, 0, 0, 0, 0, 0};
  __builtin_amdgcn_tensor_load_to_lds(g0, g1, gz4, gz4, gz8, 0);
}

// =============================================================================
// Kernel 1: per-pixel linear projections.
//   shortcut = x @ attn_W^T + attn_b   -> stored bf16 TRANSPOSED [B][C][HW]
//   emb      = x @ emb_W^T  + emb_b    -> stored bf16 row-major  [B][HW][F]
//   sq[n]    = |emb[n]|^2 ; Tinst[n] = shortcut[n].thr_W + thr_b
// One wave handles a 16-pixel tile; all GEMMs via v_wmma_f32_16x16x32_bf16.
// =============================================================================
__global__ __launch_bounds__(256)
void iam_proj_kernel(const float* __restrict__ fm,
                     const float* __restrict__ embW, const float* __restrict__ embB,
                     const float* __restrict__ attW, const float* __restrict__ attB,
                     const float* __restrict__ thrW, const float* __restrict__ thrB,
                     __bf16* __restrict__ embBF, __bf16* __restrict__ scT,
                     float* __restrict__ sqv, float* __restrict__ tinst) {
  __shared__ __bf16 sT[WPB][16][16];            // 4 KB: per-wave store-transpose

  const int lane = threadIdx.x & 31;
  const int wav  = threadIdx.x >> 5;
  const int gw   = blockIdx.x * WPB + wav;      // 0..2047 wave tiles
  const int b    = gw >> 6;
  const int n0   = (gw & 63) << 4;              // 16-pixel tile base
  const int lm   = lane & 15;
  const int hi   = lane >> 4;
  const int kb   = hi * 8;

  // A fragments: x tile, 16 rows x 256 K, 8 chunks of K=32 (f32 -> bf16)
  const float* xrow = fm + (size_t)(b * HW + n0 + lm) * CHN;
  v16bf xa[8];
#pragma unroll
  for (int kc = 0; kc < 8; ++kc) xa[kc] = frag_from_f32(xrow + kc * 32, kb);

  float tacc[8]  = {};
  float sqacc[8] = {};

  // ---- shortcut: 16 N-tiles of 16 channels --------------------------------
#pragma unroll 4
  for (int nt = 0; nt < 16; ++nt) {
    const int c = nt * 16 + lm;
    const float* wrow = attW + (size_t)c * CHN;   // B column c = attn_W row c
    v8f acc = {};
#pragma unroll
    for (int kc = 0; kc < 8; ++kc) {
      v16bf wb = frag_from_f32(wrow + kc * 32, kb);
      acc = __builtin_amdgcn_wmma_f32_16x16x32_bf16(false, xa[kc], false, wb,
                                                    (short)0, acc, false, false);
    }
    const float bias = attB[c];
    const float tw   = thrW[c];
#pragma unroll
    for (int v = 0; v < 8; ++v) {
      const float s = acc[v] + bias;              // (row n0+v+8*hi, chan c)
      tacc[v] += s * tw;
      sT[wav][lm][v + 8 * hi] = (__bf16)s;        // transpose through LDS
    }
    // each lane now stores one aligned 16B run of channel row c
    *((uint4*)(&scT[(size_t)(b * CHN + c) * HW + n0 + hi * 8])) =
        *((const uint4*)&sT[wav][lm][hi * 8]);
  }

  // ---- emb: 4 N-tiles of 16 features --------------------------------------
#pragma unroll
  for (int nt = 0; nt < 4; ++nt) {
    const int f = nt * 16 + lm;
    const float* wrow = embW + (size_t)f * CHN;
    v8f acc = {};
#pragma unroll
    for (int kc = 0; kc < 8; ++kc) {
      v16bf wb = frag_from_f32(wrow + kc * 32, kb);
      acc = __builtin_amdgcn_wmma_f32_16x16x32_bf16(false, xa[kc], false, wb,
                                                    (short)0, acc, false, false);
    }
    const float bias = embB[f];
#pragma unroll
    for (int v = 0; v < 8; ++v) {
      const float e = acc[v] + bias;
      sqacc[v] += e * e;
      const int n = n0 + v + 8 * hi;
      embBF[(size_t)(b * HW + n) * FD + f] = (__bf16)e;
    }
  }

  // reduce across the 16 lanes that share each row (butterfly over low nibble)
#pragma unroll
  for (int m = 1; m < 16; m <<= 1) {
#pragma unroll
    for (int v = 0; v < 8; ++v) {
      tacc[v]  += __shfl_xor(tacc[v],  m, 32);
      sqacc[v] += __shfl_xor(sqacc[v], m, 32);
    }
  }
  if (lm == 0) {
    const float tb = thrB[0];
#pragma unroll
    for (int v = 0; v < 8; ++v) {
      const int n = n0 + v + 8 * hi;
      tinst[b * HW + n] = tacc[v] + tb;
      sqv[b * HW + n]   = sqacc[v];
    }
  }
}

// =============================================================================
// Kernel 2: fused flash-style attention. One wave owns a 16-row i-tile;
// 8 waves/block share LDS-staged j-tiles. The 16KB shortcut^T tile is moved
// by the Tensor Data Mover (double-buffered, overlapped with compute).
//   pass1: s1[i] = sum_j exp( exp(-d2) - 1 + mask )
//   pass2: p = exp(relu(exp(z)/s1 - T_j)); s2 += p; O += P @ shortcut (WMMA)
//   out   = x + O / s2
// =============================================================================
__global__ __launch_bounds__(256)
void iam_attn_kernel(const float* __restrict__ fm,
                     const __bf16* __restrict__ embBF, const __bf16* __restrict__ scT,
                     const float* __restrict__ sqv, const float* __restrict__ tinst,
                     float* __restrict__ out) {
  __shared__ __bf16 sEmb[TJ][FD];           //  4 KB: emb rows j0..j0+31
  __shared__ __bf16 sSC[2][CHN][TJ];        // 32 KB: TDM double buffer [c][j]
  __shared__ __bf16 sP[WPB][16][TJ];        //  8 KB: per-wave P transpose buffer

  const int t    = threadIdx.x;
  const int lane = t & 31;
  const int wav  = t >> 5;
  const int b    = blockIdx.x >> 3;
  const int i0   = (((blockIdx.x & 7) * WPB) + wav) * 16;
  const int lm   = lane & 15;
  const int hi   = lane >> 4;
  const int kb   = hi * 8;

  const __bf16* gEmbB = embBF + (size_t)b * HW * FD;
  const __bf16* gSCb  = scT   + (size_t)b * CHN * HW;

  // Q fragments (K = 64 in two chunks of 32), resident for whole kernel
  const __bf16* qrow = gEmbB + (size_t)(i0 + lm) * FD;
  const v16bf qa = frag_from_bf16(qrow, kb);
  const v16bf qb = frag_from_bf16(qrow + 32, kb);

  float sqi[8]; int yi[8], xi[8];
#pragma unroll
  for (int v = 0; v < 8; ++v) {
    const int i = i0 + v + 8 * hi;
    sqi[v] = sqv[b * HW + i];
    yi[v]  = i >> 5;
    xi[v]  = i & 31;
  }

  // ---------------- pass 1: first-softmax denominator ----------------------
  float a1[8] = {};
  for (int jt = 0; jt < NJ; ++jt) {
    const int j0 = jt * TJ;
    __syncthreads();
    { // stage emb tile: 4KB, one 16B chunk per thread
      const int r = t >> 3, fo = (t & 7) * 8;
      *((uint4*)&sEmb[r][fo]) =
          *((const uint4*)(gEmbB + (size_t)(j0 + r) * FD + fo));
    }
    __syncthreads();
#pragma unroll
    for (int sub = 0; sub < 2; ++sub) {
      const int jr = sub * 16 + lm;
      const v16bf kf0 = frag_from_bf16(&sEmb[jr][0],  kb);
      const v16bf kf1 = frag_from_bf16(&sEmb[jr][32], kb);
      v8f d = {};
      d = __builtin_amdgcn_wmma_f32_16x16x32_bf16(false, qa, false, kf0, (short)0, d, false, false);
      d = __builtin_amdgcn_wmma_f32_16x16x32_bf16(false, qb, false, kf1, (short)0, d, false, false);
      const int   j   = j0 + jr;
      const float sqj = sqv[b * HW + j];
      const int   yj  = j >> 5, xj = j & 31;
#pragma unroll
      for (int v = 0; v < 8; ++v) {
        int dy = yi[v] - yj; dy = dy < 0 ? -dy : dy;
        int dx = xi[v] - xj; dx = dx < 0 ? -dx : dx;
        const int   diff  = dy + dx;
        const float maskv = (diff <= 32) ? (float)diff * (1.0f / 32.0f) : 0.0f;
        const float d2    = fmaxf(sqi[v] + sqj - 2.0f * d[v], 0.0f);
        const float z     = __expf(-d2) - 1.0f + maskv;   // z <= 1 -> exp safe
        a1[v] += __expf(z);
      }
    }
  }
#pragma unroll
  for (int m = 1; m < 16; m <<= 1)
#pragma unroll
    for (int v = 0; v < 8; ++v) a1[v] += __shfl_xor(a1[v], m, 32);
  float r1[8];
#pragma unroll
  for (int v = 0; v < 8; ++v) r1[v] = 1.0f / a1[v];

  // ---------------- pass 2: P = softmax2 numerator, O = P @ shortcut -------
  v8f   o[16];
#pragma unroll
  for (int nt = 0; nt < 16; ++nt) o[nt] = (v8f){};
  float a2[8] = {};

  const unsigned lds_sc0 = (unsigned)(unsigned long long)(void*)&sSC[0][0][0];
  const unsigned lds_sc1 = (unsigned)(unsigned long long)(void*)&sSC[1][0][0];
  if (wav == 0) { // prime the TDM pipeline: tile 0 -> buffer 0
    tdm_load_2d_bf16(gSCb, lds_sc0, TJ, CHN, HW, CHN, HW);
  }

  for (int jt = 0; jt < NJ; ++jt) {
    const int j0 = jt * TJ;
    __syncthreads();               // everyone finished consuming buffer (jt+1)&1
    { // stage emb tile (regular path, 4KB)
      const int r = t >> 3, fo = (t & 7) * 8;
      *((uint4*)&sEmb[r][fo]) =
          *((const uint4*)(gEmbB + (size_t)(j0 + r) * FD + fo));
    }
    if (jt + 1 < NJ)               // gfx1250 global_prefetch_b8 for next emb tile
      __builtin_prefetch(gEmbB + (size_t)(j0 + TJ + (t >> 3)) * FD, 0, 1);
    if (wav == 0) {
      if (jt + 1 < NJ) {           // async DMA of next tile into the other buffer
        tdm_load_2d_bf16(gSCb + (size_t)(jt + 1) * TJ,
                         (jt & 1) ? lds_sc0 : lds_sc1, TJ, CHN, HW, CHN, HW);
        __builtin_amdgcn_s_wait_tensorcnt(1);   // tile jt resident (in-order TDM)
      } else {
        __builtin_amdgcn_s_wait_tensorcnt(0);
      }
    }
    __syncthreads();
    const __bf16* scTile = &sSC[jt & 1][0][0];

#pragma unroll
    for (int sub = 0; sub < 2; ++sub) {
      const int jr = sub * 16 + lm;
      const v16bf kf0 = frag_from_bf16(&sEmb[jr][0],  kb);
      const v16bf kf1 = frag_from_bf16(&sEmb[jr][32], kb);
      v8f d = {};
      d = __builtin_amdgcn_wmma_f32_16x16x32_bf16(false, qa, false, kf0, (short)0, d, false, false);
      d = __builtin_amdgcn_wmma_f32_16x16x32_bf16(false, qb, false, kf1, (short)0, d, false, false);
      const int   j   = j0 + jr;
      const float sqj = sqv[b * HW + j];
      const float tj  = tinst[b * HW + j];
      const int   yj  = j >> 5, xj = j & 31;
#pragma unroll
      for (int v = 0; v < 8; ++v) {
        int dy = yi[v] - yj; dy = dy < 0 ? -dy : dy;
        int dx = xi[v] - xj; dx = dx < 0 ? -dx : dx;
        const int   diff  = dy + dx;
        const float maskv = (diff <= 32) ? (float)diff * (1.0f / 32.0f) : 0.0f;
        const float d2    = fmaxf(sqi[v] + sqj - 2.0f * d[v], 0.0f);
        const float z     = __expf(-d2) - 1.0f + maskv;
        const float sim   = __expf(z) * r1[v];
        const float p     = __expf(fmaxf(sim - tj, 0.0f)); // sim in [0,1] -> safe
        a2[v] += p;
        sP[wav][v + 8 * hi][sub * 16 + lm] = (__bf16)p;    // D-layout -> tile
      }
    }
    // LDS transpose: re-read P as a 16x32 bf16 A-fragment (same-wave RAW,
    // compiler inserts s_wait_dscnt); then 16 N-tile WMMAs over the channels
    const v16bf pa = frag_from_bf16(&sP[wav][lm][0], kb);
#pragma unroll 4
    for (int nt = 0; nt < 16; ++nt) {
      const v16bf sb = frag_from_bf16(scTile + (size_t)(nt * 16 + lm) * TJ, kb);
      o[nt] = __builtin_amdgcn_wmma_f32_16x16x32_bf16(false, pa, false, sb,
                                                      (short)0, o[nt], false, false);
    }
  }

#pragma unroll
  for (int m = 1; m < 16; m <<= 1)
#pragma unroll
    for (int v = 0; v < 8; ++v) a2[v] += __shfl_xor(a2[v], m, 32);
  float r2[8];
#pragma unroll
  for (int v = 0; v < 8; ++v) r2[v] = 1.0f / a2[v];

  // epilogue: out = x + O / s2
#pragma unroll 4
  for (int nt = 0; nt < 16; ++nt) {
    const int c = nt * 16 + lm;
#pragma unroll
    for (int v = 0; v < 8; ++v) {
      const int i = i0 + v + 8 * hi;
      const size_t idx = (size_t)(b * HW + i) * CHN + c;
      out[idx] = fm[idx] + o[nt][v] * r2[v];
    }
  }
}

// =============================================================================
extern "C" void kernel_launch(void* const* d_in, const int* in_sizes, int n_in,
                              void* d_out, int out_size, void* d_ws, size_t ws_size,
                              hipStream_t stream) {
  (void)in_sizes; (void)n_in; (void)out_size; (void)ws_size;
  const float* fm   = (const float*)d_in[0];
  const float* embW = (const float*)d_in[1];
  const float* embB = (const float*)d_in[2];
  const float* attW = (const float*)d_in[3];
  const float* attB = (const float*)d_in[4];
  const float* thrW = (const float*)d_in[5];
  const float* thrB = (const float*)d_in[6];
  float* out = (float*)d_out;

  char* ws = (char*)d_ws;
  const size_t off_emb = 0;                                       //  4 MiB bf16
  const size_t off_sct = off_emb + (size_t)BATCH * HW * FD * 2;   // 16 MiB bf16
  const size_t off_sq  = off_sct + (size_t)BATCH * CHN * HW * 2;  // 128 KiB f32
  const size_t off_t   = off_sq + (size_t)BATCH * HW * 4;         // 128 KiB f32
  __bf16* embBF = (__bf16*)(ws + off_emb);
  __bf16* scT   = (__bf16*)(ws + off_sct);
  float*  sqv   = (float*)(ws + off_sq);
  float*  tinst = (float*)(ws + off_t);

  // 2048 wave tiles / 8 waves per block
  iam_proj_kernel<<<256, 256, 0, stream>>>(fm, embW, embB, attW, attB, thrW, thrB,
                                           embBF, scT, sqv, tinst);
  // 32 batches x 8 row-blocks (128 rows each)
  iam_attn_kernel<<<256, 256, 0, stream>>>(fm, embBF, scT, sqv, tinst, out);
}
